// MultiHeadSelfAttention_27470610826057
// MI455X (gfx1250) — matrix operand order repaired and checked by reading, not compiled, not run on hardware
//
#include <hip/hip_runtime.h>
#include <hip/hip_bf16.h>

// ---------------------------------------------------------------------------
// MI455X (gfx1250) multi-head self-attention, bf16 WMMA path, f32 accumulate.
// B=2, T=2048, C=1024, H=16, Dh=64.
// ---------------------------------------------------------------------------

#define D_MODEL 1024
#define N_HEADS 16
#define D_HEAD  64
#define BATCH   2
#define SEQ     2048
#define NTOK    (BATCH * SEQ)          // 4096
#define QKV_N   (3 * D_MODEL)          // 3072

typedef __attribute__((ext_vector_type(16))) __bf16 bf16x16;
typedef __attribute__((ext_vector_type(8)))  __bf16 bf16x8;
typedef __attribute__((ext_vector_type(8)))  float  f32x8;
typedef __attribute__((ext_vector_type(8)))  unsigned short u16x8;

static __device__ __forceinline__ unsigned short f2bf(float f) {
    unsigned u = __builtin_bit_cast(unsigned, f);
    u += 0x7FFFu + ((u >> 16) & 1u);   // round-to-nearest-even
    return (unsigned short)(u >> 16);
}

// A-fragment (16x32 bf16, MxK): lane holds row M=lane&15; K elements
// {kb0..kb0+7, kb1..kb1+7} with (kb0,kb1) = (0,8) lanes 0-15, (16,24) lanes 16-31.
static __device__ __forceinline__ bf16x16 load_afrag(const unsigned short* p0,
                                                     const unsigned short* p1) {
    bf16x8 lo = *(const bf16x8*)p0;
    bf16x8 hi = *(const bf16x8*)p1;
    return __builtin_shufflevector(lo, hi, 0, 1, 2, 3, 4, 5, 6, 7,
                                           8, 9, 10, 11, 12, 13, 14, 15);
}

static __device__ __forceinline__ f32x8 wmma_bf16(bf16x16 a, bf16x16 b, f32x8 c) {
    return __builtin_amdgcn_wmma_f32_16x16x32_bf16(
        /*neg_a=*/false, a, /*neg_b=*/false, b,
        /*c_mod=*/(short)0, c, /*reuse_a=*/false, /*reuse_b=*/false);
}

// ---------------------------------------------------------------------------
// fp32 -> bf16 conversion
// ---------------------------------------------------------------------------
__global__ void cvt_f32_to_bf16(const float* __restrict__ in,
                                unsigned short* __restrict__ out, int n) {
    int i = blockIdx.x * blockDim.x + threadIdx.x;
    int stride = gridDim.x * blockDim.x;
    for (; i < n; i += stride) out[i] = f2bf(in[i]);
}

// ---------------------------------------------------------------------------
// QKV projection: qkv = x @ W_qkv^T, scattered to Q[B,H,T,Dh], K[B,H,T,Dh],
// Vt[B,H,Dh,T] (V transposed so PV B-fragments are contiguous).
// Block = 256 threads = 8 waves; wave tile 32x64 = 2x4 WMMA tiles; K step 32.
// The 256-col block tile sits entirely inside one of Q/K/V -> c is uniform.
// ---------------------------------------------------------------------------
__global__ __launch_bounds__(256) void qkv_gemm(
    const unsigned short* __restrict__ xb,      // [NTOK, 1024] bf16
    const unsigned short* __restrict__ wb,      // [3072, 1024] bf16
    unsigned short* __restrict__ Qb,
    unsigned short* __restrict__ Kb,
    unsigned short* __restrict__ Vt) {
    const int lane  = threadIdx.x & 31;
    const int wave  = threadIdx.x >> 5;
    const int lhalf = lane >> 4;                // 0 or 1
    const int l15   = lane & 15;
    const int kb0   = lhalf * 16;               // A-frag K offsets
    const int kb1   = lhalf * 16 + 8;
    const int kbB   = lhalf * 16;               // B-frag K offset

    const int wm = blockIdx.y * 64  + (wave >> 2) * 32;
    const int wn = blockIdx.x * 256 + (wave & 3) * 64;

    f32x8 acc[2][4];
#pragma unroll
    for (int mt = 0; mt < 2; ++mt)
#pragma unroll
        for (int nt = 0; nt < 4; ++nt) acc[mt][nt] = (f32x8){0,0,0,0,0,0,0,0};

    for (int k0 = 0; k0 < D_MODEL; k0 += 32) {
        bf16x16 a[2], b[4];
#pragma unroll
        for (int mt = 0; mt < 2; ++mt) {
            const unsigned short* ar = xb + (size_t)(wm + mt * 16 + l15) * D_MODEL + k0;
            a[mt] = load_afrag(ar + kb0, ar + kb1);
        }
#pragma unroll
        for (int nt = 0; nt < 4; ++nt) {
            const unsigned short* br = wb + (size_t)(wn + nt * 16 + l15) * D_MODEL + k0 + kbB;
            b[nt] = *(const bf16x16*)br;
        }
#pragma unroll
        for (int mt = 0; mt < 2; ++mt)
#pragma unroll
            for (int nt = 0; nt < 4; ++nt)
                acc[mt][nt] = wmma_bf16(a[mt], b[nt], acc[mt][nt]);
    }

    const int cblk = blockIdx.x >> 2;           // 0=Q 1=K 2=V, uniform per block
    if (cblk < 2) {
        unsigned short* dst = (cblk == 0) ? Qb : Kb;
#pragma unroll
        for (int mt = 0; mt < 2; ++mt) {
#pragma unroll
            for (int nt = 0; nt < 4; ++nt) {
                const int n   = wn + nt * 16 + l15;
                const int rem = n & 1023;
                const int h   = rem >> 6;
                const int d   = rem & 63;
#pragma unroll
                for (int r = 0; r < 8; ++r) {
                    const int token = wm + mt * 16 + r + lhalf * 8;
                    const int bb    = token >> 11;
                    const int t     = token & 2047;
                    const int bh    = bb * N_HEADS + h;
                    dst[((size_t)bh * SEQ + t) * D_HEAD + d] = f2bf(acc[mt][nt][r]);
                }
            }
        }
    } else {
        // V, transposed layout: lane's 8 rows are 8 consecutive t -> one b128 store
#pragma unroll
        for (int mt = 0; mt < 2; ++mt) {
#pragma unroll
            for (int nt = 0; nt < 4; ++nt) {
                const int n   = wn + nt * 16 + l15;
                const int rem = n & 1023;
                const int h   = rem >> 6;
                const int d   = rem & 63;
                const int token0 = wm + mt * 16 + lhalf * 8;   // 8-aligned
                const int bb  = token0 >> 11;
                const int t0  = token0 & 2047;
                const int bh  = bb * N_HEADS + h;
                u16x8 pk;
#pragma unroll
                for (int r = 0; r < 8; ++r) pk[r] = f2bf(acc[mt][nt][r]);
                *(u16x8*)&Vt[((size_t)bh * D_HEAD + d) * SEQ + t0] = pk;
            }
        }
    }
}

// ---------------------------------------------------------------------------
// Flash attention: grid (T/64, B*H), block 128 = 4 waves, each wave 16 q rows.
// Per 32-key step: 2x(2 WMMA) scores, online softmax (f32), P through LDS
// (C-layout -> A-layout), 4 WMMA for P@V.  Output attn[B*T, 1024] bf16.
// ---------------------------------------------------------------------------
#define PROW 40   // padded LDS row stride (bf16 elems); 80B = 16B multiple

__global__ __launch_bounds__(128) void attn_kernel(
    const unsigned short* __restrict__ Qb,
    const unsigned short* __restrict__ Kb,
    const unsigned short* __restrict__ Vt,
    unsigned short* __restrict__ attnb) {
    const int lane  = threadIdx.x & 31;
    const int wave  = threadIdx.x >> 5;
    const int lhalf = lane >> 4;
    const int l15   = lane & 15;
    const int kb0   = lhalf * 16;
    const int kb1   = lhalf * 16 + 8;
    const int kbB   = lhalf * 16;

    const int bh    = blockIdx.y;
    const int qbase = blockIdx.x * 64 + wave * 16;

    __shared__ __align__(16) unsigned short pbuf[4 * 16 * PROW];
    unsigned short* pw = pbuf + wave * 16 * PROW;

    const unsigned short* Qh = Qb + (size_t)bh * SEQ * D_HEAD;
    const unsigned short* Kh = Kb + (size_t)bh * SEQ * D_HEAD;
    const unsigned short* Vh = Vt + (size_t)bh * D_HEAD * SEQ;

    // Q fragments (held in registers for the whole KV loop)
    const unsigned short* qp = Qh + (size_t)(qbase + l15) * D_HEAD;
    const bf16x16 qa0 = load_afrag(qp + kb0,      qp + kb1);
    const bf16x16 qa1 = load_afrag(qp + 32 + kb0, qp + 32 + kb1);

    float mrow[8], lrow[8];
#pragma unroll
    for (int r = 0; r < 8; ++r) { mrow[r] = -1.0e30f; lrow[r] = 0.0f; }
    f32x8 oacc[4];
#pragma unroll
    for (int t = 0; t < 4; ++t) oacc[t] = (f32x8){0,0,0,0,0,0,0,0};

    const float sm_scale = 0.125f;  // 1/sqrt(64)

    for (int kv = 0; kv < SEQ; kv += 32) {
        // --- scores: two 16x16 subtiles over K=Dh=64 ---
        f32x8 S[2];
#pragma unroll
        for (int s = 0; s < 2; ++s) {
            const unsigned short* kp = Kh + (size_t)(kv + s * 16 + l15) * D_HEAD;
            const bf16x16 kbf0 = *(const bf16x16*)(kp + kbB);
            const bf16x16 kbf1 = *(const bf16x16*)(kp + 32 + kbB);
            f32x8 c = (f32x8){0,0,0,0,0,0,0,0};
            c = wmma_bf16(qa0, kbf0, c);
            c = wmma_bf16(qa1, kbf1, c);
            S[s] = c;
        }

        // --- online softmax per row (rows live per-VGPR across 16-lane halves) ---
#pragma unroll
        for (int r = 0; r < 8; ++r) {
            const float s0 = S[0][r] * sm_scale;
            const float s1 = S[1][r] * sm_scale;
            float tmax = fmaxf(s0, s1);
            tmax = fmaxf(tmax, __shfl_xor(tmax, 1, 32));
            tmax = fmaxf(tmax, __shfl_xor(tmax, 2, 32));
            tmax = fmaxf(tmax, __shfl_xor(tmax, 4, 32));
            tmax = fmaxf(tmax, __shfl_xor(tmax, 8, 32));
            const float mn = fmaxf(mrow[r], tmax);
            const float p0 = __expf(s0 - mn);
            const float p1 = __expf(s1 - mn);
            float ts = p0 + p1;
            ts += __shfl_xor(ts, 1, 32);
            ts += __shfl_xor(ts, 2, 32);
            ts += __shfl_xor(ts, 4, 32);
            ts += __shfl_xor(ts, 8, 32);
            const float ef = __expf(mrow[r] - mn);
            lrow[r] = lrow[r] * ef + ts;
            mrow[r] = mn;
#pragma unroll
            for (int t = 0; t < 4; ++t) oacc[t][r] *= ef;
            // stash probabilities (C-layout -> LDS 16x32 tile)
            pw[(r + lhalf * 8) * PROW + l15]      = f2bf(p0);
            pw[(r + lhalf * 8) * PROW + 16 + l15] = f2bf(p1);
        }
        __asm__ volatile("s_wait_dscnt 0" ::: "memory");

        // --- reload P in A-fragment layout, accumulate P @ V ---
        const unsigned short* pr = pw + l15 * PROW;
        const bf16x16 pa = load_afrag(pr + kb0, pr + kb1);
#pragma unroll
        for (int t = 0; t < 4; ++t) {
            const unsigned short* vp =
                Vh + (size_t)(t * 16 + l15) * SEQ + kv + kbB;
            const bf16x16 vb = *(const bf16x16*)vp;
            oacc[t] = wmma_bf16(pa, vb, oacc[t]);
        }
        __asm__ volatile("" ::: "memory");  // keep LDS read before next-iter writes
    }

    // --- normalize and store to attn[B*T, C] with col = h*64 + dh ---
    const int b  = bh >> 4;
    const int h  = bh & 15;
#pragma unroll
    for (int t = 0; t < 4; ++t) {
#pragma unroll
        for (int r = 0; r < 8; ++r) {
            const int row   = qbase + r + lhalf * 8;
            const int token = b * SEQ + row;
            const int col   = h * D_HEAD + t * 16 + l15;
            attnb[(size_t)token * D_MODEL + col] = f2bf(oacc[t][r] / lrow[r]);
        }
    }
}

// ---------------------------------------------------------------------------
// Output projection: out = attn @ W_out^T  (f32 output)
// ---------------------------------------------------------------------------
__global__ __launch_bounds__(256) void out_gemm(
    const unsigned short* __restrict__ ab,      // [NTOK, 1024] bf16
    const unsigned short* __restrict__ wb,      // [1024, 1024] bf16
    float* __restrict__ out) {
    const int lane  = threadIdx.x & 31;
    const int wave  = threadIdx.x >> 5;
    const int lhalf = lane >> 4;
    const int l15   = lane & 15;
    const int kb0   = lhalf * 16;
    const int kb1   = lhalf * 16 + 8;
    const int kbB   = lhalf * 16;

    const int wm = blockIdx.y * 64  + (wave >> 2) * 32;
    const int wn = blockIdx.x * 256 + (wave & 3) * 64;

    f32x8 acc[2][4];
#pragma unroll
    for (int mt = 0; mt < 2; ++mt)
#pragma unroll
        for (int nt = 0; nt < 4; ++nt) acc[mt][nt] = (f32x8){0,0,0,0,0,0,0,0};

    for (int k0 = 0; k0 < D_MODEL; k0 += 32) {
        bf16x16 a[2], b[4];
#pragma unroll
        for (int mt = 0; mt < 2; ++mt) {
            const unsigned short* ar = ab + (size_t)(wm + mt * 16 + l15) * D_MODEL + k0;
            a[mt] = load_afrag(ar + kb0, ar + kb1);
        }
#pragma unroll
        for (int nt = 0; nt < 4; ++nt) {
            const unsigned short* br = wb + (size_t)(wn + nt * 16 + l15) * D_MODEL + k0 + kbB;
            b[nt] = *(const bf16x16*)br;
        }
#pragma unroll
        for (int mt = 0; mt < 2; ++mt)
#pragma unroll
            for (int nt = 0; nt < 4; ++nt)
                acc[mt][nt] = wmma_bf16(a[mt], b[nt], acc[mt][nt]);
    }

#pragma unroll
    for (int mt = 0; mt < 2; ++mt)
#pragma unroll
        for (int nt = 0; nt < 4; ++nt) {
            const int n = wn + nt * 16 + l15;
#pragma unroll
            for (int r = 0; r < 8; ++r) {
                const int m = wm + mt * 16 + r + lhalf * 8;
                out[(size_t)m * D_MODEL + n] = acc[mt][nt][r];
            }
        }
}

// ---------------------------------------------------------------------------
// Host-side launcher
// ---------------------------------------------------------------------------
extern "C" void kernel_launch(void* const* d_in, const int* in_sizes, int n_in,
                              void* d_out, int out_size, void* d_ws, size_t ws_size,
                              hipStream_t stream) {
    (void)in_sizes; (void)n_in; (void)out_size; (void)ws_size;

    const float* x     = (const float*)d_in[0];   // [2,2048,1024]
    const float* W_qkv = (const float*)d_in[1];   // [3072,1024]
    const float* W_out = (const float*)d_in[2];   // [1024,1024]
    float* out = (float*)d_out;                   // [2,2048,1024]

    char* ws = (char*)d_ws;
    const size_t n_x    = (size_t)NTOK * D_MODEL;        // 4,194,304
    const size_t n_wqkv = (size_t)QKV_N * D_MODEL;       // 3,145,728
    const size_t n_wout = (size_t)D_MODEL * D_MODEL;     // 1,048,576

    unsigned short* xb    = (unsigned short*)(ws);                       // 8 MB
    unsigned short* wqkvb = (unsigned short*)(ws + 8388608);             // 6 MB
    unsigned short* woutb = (unsigned short*)(ws + 14680064);            // 2 MB
    unsigned short* Qb    = (unsigned short*)(ws + 16777216);            // 8 MB
    unsigned short* Kb    = (unsigned short*)(ws + 25165824);            // 8 MB
    unsigned short* Vt    = (unsigned short*)(ws + 33554432);            // 8 MB
    unsigned short* attnb = (unsigned short*)(ws + 41943040);            // 8 MB

    cvt_f32_to_bf16<<<4096, 256, 0, stream>>>(x, xb, (int)n_x);
    cvt_f32_to_bf16<<<4096, 256, 0, stream>>>(W_qkv, wqkvb, (int)n_wqkv);
    cvt_f32_to_bf16<<<2048, 256, 0, stream>>>(W_out, woutb, (int)n_wout);

    // QKV: M=4096 (y: 64 tiles of 64), N=3072 (x: 12 tiles of 256)
    qkv_gemm<<<dim3(12, 64), 256, 0, stream>>>(xb, wqkvb, Qb, Kb, Vt);

    // Attention: x = T/64 q-tiles, y = B*H
    attn_kernel<<<dim3(SEQ / 64, BATCH * N_HEADS), 128, 0, stream>>>(Qb, Kb, Vt, attnb);

    // Out proj: M=4096 (y: 64), N=1024 (x: 4 tiles of 256)
    out_gemm<<<dim3(4, 64), 256, 0, stream>>>(attnb, woutb, out);
}
